// TemporalMSA_59605556134026
// MI455X (gfx1250) — matrix operand order repaired and checked by reading, not compiled, not run on hardware
//
#include <hip/hip_runtime.h>
#include <hip/hip_bf16.h>
#include <math.h>

typedef _Float16 half_t;
typedef __attribute__((ext_vector_type(16))) _Float16 v16h;
typedef __attribute__((ext_vector_type(8)))  _Float16 v8h;
typedef __attribute__((ext_vector_type(8)))  float    v8f;
typedef __attribute__((ext_vector_type(4)))  unsigned int u32x4;
typedef __attribute__((ext_vector_type(8)))  int          i32x8;
typedef __attribute__((ext_vector_type(4)))  int          i32x4;

#define T_DIM      16
#define C_DIM      128
#define HW         3136     // 56*56
#define SEQ_PER_WG 8
#define NTHREADS   256

#define WQ_HALVES  (3 * C_DIM * C_DIM)   // 49152
#define WP_HALVES  (C_DIM * C_DIM)       // 16384
#define W_HALVES   (WQ_HALVES + WP_HALVES) // 65536 halves = 128KB contiguous in d_ws

// ---- LDS byte offsets (dynamic shared; total 266240 B < 320KB/WGP) ----
#define OFF_WQKV  0u        // 384*128 f16 = 98304 B   (wproj follows contiguously)
#define OFF_WPROJ 98304u    // 128*128 f16 = 32768 B
#define OFF_XY    131072u   // 8 seq * 16*128 f16 = 32768 B  (X, later reused for Y)
#define OFF_QK    163840u   // 8 seq * 16*256 f16 = 65536 B  (Q|K, later reused as f32 out)
#define OFF_VT    229376u   // 8 seq * 128*16 f16 = 32768 B  (V transposed [d][t])
#define OFF_P     262144u   // 8 seq * 16*16 f16 = 4096 B
#define LDS_BYTES 266240u

static __device__ inline v8f wmma_f16(v16h a, v16h b, v8f c) {
  return __builtin_amdgcn_wmma_f32_16x16x32_f16(false, a, false, b, (short)0, c, false, false);
}

// A-matrix 16x32 f16 fragment (ISA 7.12.2): lane m = L&15; lanes>=16 offset K by +8.
static __device__ inline v16h load_a_frag(const half_t* base, int rowStride, int lane) {
  int m  = lane & 15;
  int ko = (lane >> 4) << 3;
  v8h lo = *(const v8h*)(base + m * rowStride + ko);
  v8h hi = *(const v8h*)(base + m * rowStride + ko + 16);
  v16h r;
#pragma unroll
  for (int i = 0; i < 8; ++i) { r[i] = lo[i]; r[i + 8] = hi[i]; }
  return r;
}

// B-matrix 32x16 f16 fragment: lane holds column n = L&15; K = 8*(L>>4)..+7 and +16..+23,
// matching the A-fragment k order (B[k][n] = W[n][k], weights row-major).
static __device__ inline v16h load_b_frag(const half_t* base, int nRowStride, int lane) {
  int n  = lane & 15;
  int ko = (lane >> 4) << 3;
  v8h lo = *(const v8h*)(base + n * nRowStride + ko);
  v8h hi = *(const v8h*)(base + n * nRowStride + ko + 16);
  v16h r;
#pragma unroll
  for (int i = 0; i < 8; ++i) { r[i] = lo[i]; r[i + 8] = hi[i]; }
  return r;
}

// ---- prep: convert both weight matrices f32 -> f16 once into d_ws ----
__global__ void __launch_bounds__(NTHREADS)
weights_to_f16_kernel(const float* __restrict__ wqkv,
                      const float* __restrict__ wproj,
                      half_t* __restrict__ ws) {
  int i = blockIdx.x * NTHREADS + threadIdx.x;
  if (i < WQ_HALVES)      ws[i] = (half_t)wqkv[i];
  else if (i < W_HALVES)  ws[i] = (half_t)wproj[i - WQ_HALVES];
}

__global__ void __launch_bounds__(NTHREADS)
temporal_msa_kernel(const float* __restrict__ xg,
                    const half_t* __restrict__ wf16,   // pre-converted [wqkv | wproj]
                    const float* __restrict__ bproj,
                    float* __restrict__ outg) {
  extern __shared__ char smem[];
  const int tid  = threadIdx.x;
  const int lane = tid & 31;
  const int seq  = tid >> 5;          // wave id == sequence slot 0..7

  half_t* wq_l = (half_t*)(smem + OFF_WQKV);
  half_t* wp_l = (half_t*)(smem + OFF_WPROJ);

  // ---- block -> (b, hh, ww0): n0 = b*3136 + hh*56 + ww0, 8 | 56 keeps row intact
  const int n0  = blockIdx.x * SEQ_PER_WG;
  const int b   = n0 / HW;
  const int rem = n0 % HW;
  const int hh  = rem / 56;
  const int ww0 = rem % 56;
  const size_t gbase = (size_t)b * T_DIM * C_DIM * HW + (size_t)hh * 56 + ww0;

  // ---- stage pre-converted f16 weights into LDS ----
#if __has_builtin(__builtin_amdgcn_tensor_load_to_lds)
  // Tensor Data Mover: one wave issues a single 128KB DMA (8 rows x 8192 halves),
  // overlapping with the other waves' x staging. D# per ISA sec 8.3/8.4.
  // This toolchain exposes the 6-arg form: (g0 x4, g1 x8, g2 x4, g3 x4, g4 x8, cpol).
  if (seq == 0) {
    unsigned long long ga = (unsigned long long)(uintptr_t)wf16;
    u32x4 g0;
    g0[0] = 1u;                                            // count=1, user mode
    g0[1] = OFF_WQKV;                                      // lds_addr (dyn LDS base = 0)
    g0[2] = (unsigned)(ga & 0xFFFFFFFFu);                  // global_addr low 32
    g0[3] = (unsigned)((ga >> 32) & 0x01FFFFFFu) | (2u << 30); // addr hi | type=2 (image)
    i32x8 g1;
    g1[0] = 0x00010000;            // workgroup_mask=0, data_size=1 (2 bytes)
    g1[1] = (int)(8192u << 16);    // tensor_dim0[15:0] = 8192 at bits 63:48
    g1[2] = (int)(8u << 16);       // tensor_dim1[15:0] = 8    at bits 95:80
    g1[3] = (int)(8192u << 16);    // tile_dim0 = 8192         at bits 127:112
    g1[4] = 8;                     // tile_dim1 = 8
    g1[5] = 8192;                  // tensor_dim0_stride = 8192 (low 32)
    g1[6] = 0;
    g1[7] = 0;
    i32x4 g2 = {0, 0, 0, 0};       // 2D copy: dims 2/3 unused
    i32x4 g3 = {0, 0, 0, 0};
    i32x8 g4 = {0, 0, 0, 0, 0, 0, 0, 0};   // trailing group unused for 2D copy
    __builtin_amdgcn_tensor_load_to_lds(g0, g1, g2, g3, g4, 0);
    __builtin_amdgcn_s_wait_tensorcnt(0);
  }
#else
  {
    const v8h* src = (const v8h*)wf16;
    v8h* dst = (v8h*)(smem + OFF_WQKV);
    for (int i = tid; i < W_HALVES / 8; i += NTHREADS) dst[i] = src[i];
  }
#endif

  // ---- cooperative load: X tile, 8 consecutive ww per (t,c) => 32B runs
  {
    half_t* xl = (half_t*)(smem + OFF_XY);
#pragma unroll 4
    for (int i = tid; i < SEQ_PER_WG * T_DIM * C_DIM; i += NTHREADS) {
      int s  = i & 7;       // ww offset
      int tc = i >> 3;      // t*128 + c
      xl[s * (T_DIM * C_DIM) + tc] = (half_t)xg[gbase + (size_t)tc * HW + s];
    }
  }
  __syncthreads();

  // per-wave LDS regions
  half_t* xy  = (half_t*)(smem + OFF_XY) + seq * (T_DIM * C_DIM);   // X, later Y
  half_t* qk  = (half_t*)(smem + OFF_QK) + seq * (T_DIM * 256);     // Q(0..127)|K(128..255)
  half_t* vT  = (half_t*)(smem + OFF_VT) + seq * (C_DIM * T_DIM);   // V^T [d][t]
  half_t* pbf = (half_t*)(smem + OFF_P)  + seq * (T_DIM * T_DIM);   // softmax P [t][s]
  float*  obf = (float*)(smem + OFF_QK)  + seq * (T_DIM * C_DIM);   // f32 out (reuses Q|K)

  const int scol = lane & 15;            // N column / s index
  const int thi  = (lane >> 4) << 3;     // +8 row offset for upper half-wave

  // ================= Phase 1: QKV = X[16x128] @ Wqkv^T  (96 WMMAs) =================
  v16h xa[4];
#pragma unroll
  for (int kk = 0; kk < 4; ++kk) xa[kk] = load_a_frag(xy + 32 * kk, C_DIM, lane);

  for (int j = 0; j < 24; ++j) {          // o tile: Q j=0..7, K j=8..15, V j=16..23
    v8f acc = {};
#pragma unroll
    for (int kk = 0; kk < 4; ++kk) {
      v16h wb = load_b_frag(wq_l + (16 * j) * C_DIM + 32 * kk, C_DIM, lane);
      acc = wmma_f16(xa[kk], wb, acc);
    }
    int o = 16 * j + scol;
    if (o < 256) {
#pragma unroll
      for (int r = 0; r < 8; ++r) qk[(r + thi) * 256 + o] = (half_t)acc[r];
    } else {
      int d = o - 256;
#pragma unroll
      for (int r = 0; r < 8; ++r) vT[d * T_DIM + (r + thi)] = (half_t)acc[r];
    }
  }

  // ================= Phase 2: masked-window attention per head =================
  v16h vzero;
#pragma unroll
  for (int i = 0; i < 16; ++i) vzero[i] = (half_t)0;

  for (int hd = 0; hd < 4; ++hd) {
    // S = Q_h[16x32] @ K_h^T  (K-dim = d = 32, single WMMA)
    v16h qa = load_a_frag(qk + 32 * hd, 256, lane);
    v16h kb = load_b_frag(qk + 128 + 32 * hd, 256, lane);  // B[k=d][n=s] = K[s][d]
    v8f sZ = {};
    v8f s  = wmma_f16(qa, kb, sZ);

    // scale + causal/band mask + softmax across s (16 lanes of each half-wave)
    float p[8];
#pragma unroll
    for (int r = 0; r < 8; ++r) {
      int t = r + thi;
      bool keep = (scol <= t) && ((t - scol) <= 2);  // WINDOW_T//2 = 2
      p[r] = keep ? (s[r] * 0.17677669529663689f) : -1.0e30f;
    }
#pragma unroll
    for (int r = 0; r < 8; ++r) {
      float m = p[r];
#pragma unroll
      for (int off = 8; off; off >>= 1) m = fmaxf(m, __shfl_xor(m, off));
      float e = __expf(p[r] - m);
      float sum = e;
#pragma unroll
      for (int off = 8; off; off >>= 1) sum += __shfl_xor(sum, off);
      p[r] = e / sum;
      pbf[(r + thi) * T_DIM + scol] = (half_t)p[r];
    }

    // Y_h = P[16x16 padded to K=32] @ V_h ; A upper K half zeroed, B rows>=16 zeroed
    v16h pa;
    {
      int m = lane & 15;
      v8h lo = *(const v8h*)(pbf + m * T_DIM + ((lane >> 4) << 3));
#pragma unroll
      for (int i = 0; i < 8; ++i) { pa[i] = lo[i]; pa[i + 8] = (half_t)0; }
    }
#pragma unroll
    for (int jj = 0; jj < 2; ++jj) {
      int d = 32 * hd + 16 * jj + scol;
      v16h vb = vzero;
      {
        v8h vlo = *(const v8h*)(vT + d * T_DIM + ((lane >> 4) << 3));
#pragma unroll
        for (int i = 0; i < 8; ++i) vb[i] = vlo[i];
      }
      v8f yZ = {};
      v8f y  = wmma_f16(pa, vb, yZ);
#pragma unroll
      for (int r = 0; r < 8; ++r) xy[(r + thi) * C_DIM + d] = (half_t)y[r];  // X -> Y
    }
  }

  // ================= Phase 3: out = Y[16x128] @ Wproj^T + b  (32 WMMAs) =================
  v16h ya[4];
#pragma unroll
  for (int kk = 0; kk < 4; ++kk) ya[kk] = load_a_frag(xy + 32 * kk, C_DIM, lane);

  for (int j = 0; j < 8; ++j) {
    v8f acc = {};
#pragma unroll
    for (int kk = 0; kk < 4; ++kk) {
      v16h wb = load_b_frag(wp_l + (16 * j) * C_DIM + 32 * kk, C_DIM, lane);
      acc = wmma_f16(ya[kk], wb, acc);
    }
    float bias = bproj[16 * j + scol];
#pragma unroll
    for (int r = 0; r < 8; ++r) obf[(r + thi) * C_DIM + 16 * j + scol] = acc[r] + bias;
  }

  // ================= Phase 4: cooperative coalesced writeout =================
  __syncthreads();
  const float* ol = (const float*)(smem + OFF_QK);
#pragma unroll 4
  for (int i = tid; i < SEQ_PER_WG * T_DIM * C_DIM; i += NTHREADS) {
    int s  = i & 7;
    int tc = i >> 3;
    outg[gbase + (size_t)tc * HW + s] = ol[s * (T_DIM * C_DIM) + tc];
  }
}

extern "C" void kernel_launch(void* const* d_in, const int* in_sizes, int n_in,
                              void* d_out, int out_size, void* d_ws, size_t ws_size,
                              hipStream_t stream) {
  (void)in_sizes; (void)n_in; (void)ws_size; (void)out_size;
  const float* x     = (const float*)d_in[0];
  const float* wqkv  = (const float*)d_in[1];
  const float* wproj = (const float*)d_in[2];
  const float* bprj  = (const float*)d_in[3];
  float* out   = (float*)d_out;
  half_t* wf16 = (half_t*)d_ws;   // 128KB of scratch for f16 weights

  (void)hipFuncSetAttribute((const void*)temporal_msa_kernel,
                            hipFuncAttributeMaxDynamicSharedMemorySize,
                            (int)LDS_BYTES);

  // 1) one-shot weight conversion into workspace (L2-resident thereafter)
  weights_to_f16_kernel<<<dim3(W_HALVES / NTHREADS), dim3(NTHREADS), 0, stream>>>(
      wqkv, wproj, wf16);

  // 2) fused attention kernel: 1568 WGs, 8 waves = 8 sequences each
  const int nSeq = 4 * HW;                       // B*H*W = 12544
  temporal_msa_kernel<<<dim3(nSeq / SEQ_PER_WG), dim3(NTHREADS), LDS_BYTES, stream>>>(
      x, wf16, bprj, out);
}
